// MultiHeadAttention_37349035606739
// MI455X (gfx1250) — compile-verified, matrix operand order checked by reading, not run on hardware
//
#include <hip/hip_runtime.h>
#include <hip/hip_bf16.h>

typedef __attribute__((ext_vector_type(16))) __bf16 bf16x16;
typedef __attribute__((ext_vector_type(8)))  __bf16 bf16x8;
typedef __attribute__((ext_vector_type(8)))  float  f32x8;
typedef __attribute__((ext_vector_type(4)))  int    v4i;

#define DM   1024
#define NH   16
#define DK   64
#define SEQ  2048
#define BATCH 4

#if __has_builtin(__builtin_amdgcn_global_load_async_to_lds_b128)
#define HAVE_ASYNC_LDS 1
#else
#define HAVE_ASYNC_LDS 0
#endif

__device__ __forceinline__ __bf16 f2bf(float f) { return (__bf16)f; }

union FragBF { bf16x16 v; bf16x8 h[2]; };

// 16-byte global->LDS copy: async (ASYNCcnt-tracked) if available, else via regs
__device__ __forceinline__ void cp16(void* lds, const void* g) {
#if HAVE_ASYNC_LDS
    __builtin_amdgcn_global_load_async_to_lds_b128(
        (__attribute__((address_space(1))) v4i*)g,
        (__attribute__((address_space(3))) v4i*)lds, 0, 0);
#else
    *(bf16x8*)lds = *(const bf16x8*)g;
#endif
}

// ---------------------------------------------------------------------------
// One-shot fp32 -> bf16 conversion (8 elements / thread)
// ---------------------------------------------------------------------------
__global__ __launch_bounds__(256)
void cvt_bf16(const float* __restrict__ src, __bf16* __restrict__ dst)
{
    const size_t i = ((size_t)blockIdx.x * 256 + threadIdx.x) * 8;
    const float4* s = (const float4*)(src + i);
    const float4 a = s[0], b = s[1];
    bf16x8 o;
    o[0] = f2bf(a.x); o[1] = f2bf(a.y); o[2] = f2bf(a.z); o[3] = f2bf(a.w);
    o[4] = f2bf(b.x); o[5] = f2bf(b.y); o[6] = f2bf(b.z); o[7] = f2bf(b.w);
    *(bf16x8*)(dst + i) = o;
}

// ---------------------------------------------------------------------------
// bf16 GEMM: (A[8192,1024] @ W[1024,1024]^T + bias) * oscale
//   mode 0: bf16 out, [bh][s][dk]      (Q / K)
//   mode 1: bf16 out, [bh][dk][s]      (V transposed)
//   mode 2: fp32 out, row-major [m][n] (final projection)
// Double-buffered async global->LDS staging feeding v_wmma_f32_16x16x32_bf16.
// ---------------------------------------------------------------------------
__global__ __launch_bounds__(256)
void gemm_bf16(const __bf16* __restrict__ A, const __bf16* __restrict__ W,
               const float* __restrict__ bias, __bf16* __restrict__ OutBf,
               float* __restrict__ OutF, int mode, float oscale)
{
    __shared__ __align__(16) __bf16 sA[2][64][32];
    __shared__ __align__(16) __bf16 sB[2][128][32];

    const int tid  = threadIdx.x;
    const int lane = tid & 31, l16 = lane & 15, half = (lane >> 4) & 1;
    const int wave = tid >> 5, wm = wave >> 2, wn = wave & 3;
    const int m0 = blockIdx.x * 64, n0 = blockIdx.y * 128;

    f32x8 acc[2][2] = {};

    const int ar = tid >> 2, ac = (tid & 3) * 8;     // A: 8 bf16/thread
    const int br = tid >> 1, bc = (tid & 1) * 16;    // B: 16 bf16/thread

    auto stage = [&](int buf, int k0) {
        cp16(&sA[buf][ar][ac],     A + (size_t)(m0 + ar) * DM + k0 + ac);
        cp16(&sB[buf][br][bc],     W + (size_t)(n0 + br) * DM + k0 + bc);
        cp16(&sB[buf][br][bc + 8], W + (size_t)(n0 + br) * DM + k0 + bc + 8);
    };

    stage(0, 0);
    int buf = 0;
    for (int k0 = 0; k0 < DM; k0 += 32, buf ^= 1) {
        const bool more = (k0 + 32 < DM);
        if (more) stage(buf ^ 1, k0 + 32);           // prefetch next tile
#if HAVE_ASYNC_LDS
        if (more) asm volatile("s_wait_asynccnt 0x3" ::: "memory");
        else      asm volatile("s_wait_asynccnt 0x0" ::: "memory");
#endif
        __syncthreads();

        FragBF aF[2], bF[2];
        #pragma unroll
        for (int t = 0; t < 2; ++t) {
            const int arow = wm * 32 + t * 16 + l16;
            aF[t].h[0] = *(const bf16x8*)&sA[buf][arow][half * 8];
            aF[t].h[1] = *(const bf16x8*)&sA[buf][arow][16 + half * 8];
            const int brow = wn * 32 + t * 16 + l16;
            bF[t].h[0] = *(const bf16x8*)&sB[buf][brow][half * 8];
            bF[t].h[1] = *(const bf16x8*)&sB[buf][brow][16 + half * 8];
        }
        #pragma unroll
        for (int tm = 0; tm < 2; ++tm)
            #pragma unroll
            for (int tn = 0; tn < 2; ++tn)
                acc[tm][tn] = __builtin_amdgcn_wmma_f32_16x16x32_bf16(
                    false, aF[tm].v, false, bF[tn].v, (short)0, acc[tm][tn],
                    false, false);
        __syncthreads();                              // tile fully consumed
    }

    #pragma unroll
    for (int tn = 0; tn < 2; ++tn) {
        const int gn = n0 + wn * 32 + tn * 16 + l16;
        const float bv = bias[gn];
        const int hh = gn >> 6, dd = gn & (DK - 1);
        #pragma unroll
        for (int tm = 0; tm < 2; ++tm) {
            #pragma unroll
            for (int v = 0; v < 8; ++v) {
                const int gm = m0 + wm * 32 + tm * 16 + v + 8 * half;
                const float val = (acc[tm][tn][v] + bv) * oscale;
                if (mode == 2) {
                    OutF[(size_t)gm * DM + gn] = val;
                } else {
                    const int bb = gm >> 11, ss = gm & (SEQ - 1);
                    const size_t idx = (mode == 0)
                        ? (((size_t)(bb * NH + hh) * SEQ + ss) * DK + dd)
                        : (((size_t)(bb * NH + hh) * DK + dd) * SEQ + ss);
                    OutBf[idx] = f2bf(val);
                }
            }
        }
    }
}

// ---------------------------------------------------------------------------
// Flash attention: Q[bh][s][dk] (pre-scaled), K[bh][s][dk], Vt[bh][dk][s]
// -> O written as [b][s][h*64+d] bf16 (row-major [8192,1024])
// ---------------------------------------------------------------------------
__global__ __launch_bounds__(128)
void attn(const __bf16* __restrict__ Qb, const __bf16* __restrict__ Kb,
          const __bf16* __restrict__ Vt, __bf16* __restrict__ Ob)
{
    __shared__ __align__(16) __bf16 sP[4][16][32];

    const int tid  = threadIdx.x;
    const int lane = tid & 31, l16 = lane & 15, half = (lane >> 4) & 1;
    const int wave = tid >> 5;
    const int bh = blockIdx.y;
    const int q0 = blockIdx.x * 64 + wave * 16;

    const __bf16* qrow = Qb + ((size_t)bh * SEQ + q0 + l16) * DK;
    FragBF fQ0, fQ1;
    fQ0.h[0] = *(const bf16x8*)(qrow + half * 8);
    fQ0.h[1] = *(const bf16x8*)(qrow + 16 + half * 8);
    fQ1.h[0] = *(const bf16x8*)(qrow + 32 + half * 8);
    fQ1.h[1] = *(const bf16x8*)(qrow + 48 + half * 8);

    float mrun[8], lrun[8];
    f32x8 Oacc[4] = {};
    #pragma unroll
    for (int v = 0; v < 8; ++v) { mrun[v] = -1.0e30f; lrun[v] = 0.0f; }

    for (int kv0 = 0; kv0 < SEQ; kv0 += 32) {
        f32x8 sc[2];
        #pragma unroll
        for (int j = 0; j < 2; ++j) {
            const __bf16* krow = Kb + ((size_t)bh * SEQ + kv0 + j * 16 + l16) * DK;
            FragBF fK0, fK1;
            fK0.h[0] = *(const bf16x8*)(krow + half * 8);
            fK0.h[1] = *(const bf16x8*)(krow + 16 + half * 8);
            fK1.h[0] = *(const bf16x8*)(krow + 32 + half * 8);
            fK1.h[1] = *(const bf16x8*)(krow + 48 + half * 8);
            f32x8 s = {};
            s = __builtin_amdgcn_wmma_f32_16x16x32_bf16(false, fQ0.v, false, fK0.v,
                                                        (short)0, s, false, false);
            s = __builtin_amdgcn_wmma_f32_16x16x32_bf16(false, fQ1.v, false, fK1.v,
                                                        (short)0, s, false, false);
            sc[j] = s;
        }

        #pragma unroll
        for (int v = 0; v < 8; ++v) {
            float mx = fmaxf(sc[0][v], sc[1][v]);
            mx = fmaxf(mx, __shfl_xor(mx, 1));
            mx = fmaxf(mx, __shfl_xor(mx, 2));
            mx = fmaxf(mx, __shfl_xor(mx, 4));
            mx = fmaxf(mx, __shfl_xor(mx, 8));
            const float mnew = fmaxf(mrun[v], mx);
            const float corr = __expf(mrun[v] - mnew);
            const float p0 = __expf(sc[0][v] - mnew);
            const float p1 = __expf(sc[1][v] - mnew);
            float rs = p0 + p1;
            rs += __shfl_xor(rs, 1);
            rs += __shfl_xor(rs, 2);
            rs += __shfl_xor(rs, 4);
            rs += __shfl_xor(rs, 8);
            lrun[v] = lrun[v] * corr + rs;
            mrun[v] = mnew;
            #pragma unroll
            for (int tn = 0; tn < 4; ++tn) Oacc[tn][v] *= corr;
            const int r = v + 8 * half;
            sP[wave][r][l16]      = f2bf(p0);
            sP[wave][r][16 + l16] = f2bf(p1);
        }
        asm volatile("s_wait_dscnt 0" ::: "memory");   // C-layout -> A-layout bounce

        FragBF fP;
        fP.h[0] = *(const bf16x8*)&sP[wave][l16][half * 8];
        fP.h[1] = *(const bf16x8*)&sP[wave][l16][16 + half * 8];

        #pragma unroll
        for (int tn = 0; tn < 4; ++tn) {
            const __bf16* vrow = Vt + ((size_t)bh * DK + tn * 16 + l16) * SEQ + kv0;
            FragBF fV;
            fV.h[0] = *(const bf16x8*)(vrow + half * 8);
            fV.h[1] = *(const bf16x8*)(vrow + 16 + half * 8);
            Oacc[tn] = __builtin_amdgcn_wmma_f32_16x16x32_bf16(
                false, fP.v, false, fV.v, (short)0, Oacc[tn], false, false);
        }
    }

    const int bb = bh / NH, hh = bh % NH;
    #pragma unroll
    for (int v = 0; v < 8; ++v) {
        const float inv = 1.0f / lrun[v];
        const int qg = q0 + v + 8 * half;
        #pragma unroll
        for (int tn = 0; tn < 4; ++tn) {
            const size_t idx =
                ((size_t)(bb * SEQ + qg)) * DM + hh * DK + tn * 16 + l16;
            Ob[idx] = f2bf(Oacc[tn][v] * inv);
        }
    }
}

// ---------------------------------------------------------------------------
extern "C" void kernel_launch(void* const* d_in, const int* in_sizes, int n_in,
                              void* d_out, int out_size, void* d_ws, size_t ws_size,
                              hipStream_t stream)
{
    const float* query = (const float*)d_in[0];
    const float* key_  = (const float*)d_in[1];
    const float* value = (const float*)d_in[2];
    const float* Wq = (const float*)d_in[3];
    const float* bq = (const float*)d_in[4];
    const float* Wk = (const float*)d_in[5];
    const float* bk = (const float*)d_in[6];
    const float* Wv = (const float*)d_in[7];
    const float* bv = (const float*)d_in[8];
    const float* Wo = (const float*)d_in[9];
    const float* bo = (const float*)d_in[10];

    const size_t nX = (size_t)BATCH * SEQ * DM;      // 8,388,608
    const size_t nW = (size_t)DM * DM;               // 1,048,576
    __bf16* base = (__bf16*)d_ws;
    __bf16* Xq  = base;                              // 16 MB
    __bf16* Xk  = Xq  + nX;
    __bf16* Xv  = Xk  + nX;
    __bf16* Wqb = Xv  + nX;                          // 2 MB each
    __bf16* Wkb = Wqb + nW;
    __bf16* Wvb = Wkb + nW;
    __bf16* Wob = Wvb + nW;
    __bf16* Qb  = Wob + nW;                          // 16 MB each
    __bf16* Kb  = Qb  + nX;
    __bf16* Vt  = Kb  + nX;
    __bf16* Ob  = Xq;                                // reuse: Xq dead after Q proj

    // fp32 -> bf16 packing (one-shot; removes per-tile conversion from GEMMs)
    cvt_bf16<<<(unsigned)(nX / 2048), 256, 0, stream>>>(query, Xq);
    cvt_bf16<<<(unsigned)(nX / 2048), 256, 0, stream>>>(key_,  Xk);
    cvt_bf16<<<(unsigned)(nX / 2048), 256, 0, stream>>>(value, Xv);
    cvt_bf16<<<(unsigned)(nW / 2048), 256, 0, stream>>>(Wq, Wqb);
    cvt_bf16<<<(unsigned)(nW / 2048), 256, 0, stream>>>(Wk, Wkb);
    cvt_bf16<<<(unsigned)(nW / 2048), 256, 0, stream>>>(Wv, Wvb);
    cvt_bf16<<<(unsigned)(nW / 2048), 256, 0, stream>>>(Wo, Wob);

    const dim3 gGemm((BATCH * SEQ) / 64, DM / 128);  // 128 x 8 blocks
    const float qscale = 0.125f;                     // 1/sqrt(64) folded into Q

    gemm_bf16<<<gGemm, 256, 0, stream>>>(Xq, Wqb, bq, Qb, nullptr, 0, qscale);
    gemm_bf16<<<gGemm, 256, 0, stream>>>(Xk, Wkb, bk, Kb, nullptr, 0, 1.0f);
    gemm_bf16<<<gGemm, 256, 0, stream>>>(Xv, Wvb, bv, Vt, nullptr, 1, 1.0f);

    attn<<<dim3(SEQ / 64, BATCH * NH), 128, 0, stream>>>(Qb, Kb, Vt, Ob);

    gemm_bf16<<<gGemm, 256, 0, stream>>>(Ob, Wob, bo, nullptr, (float*)d_out, 2, 1.0f);
}
